// CrossNonLocal_Fusion_75685913690253
// MI455X (gfx1250) — matrix operand order repaired and checked by reading, not compiled, not run on hardware
//
#include <hip/hip_runtime.h>
#include <hip/hip_bf16.h>

typedef __bf16 bf16_t;
typedef __attribute__((ext_vector_type(16))) __bf16 v16bf;
typedef __attribute__((ext_vector_type(8)))  float   v8f;
typedef __attribute__((ext_vector_type(4)))  unsigned u32x4;
typedef __attribute__((ext_vector_type(8)))  int      i32x8;
typedef __attribute__((ext_vector_type(4)))  int      i32x4;

union Frag16 { v16bf v; uint4 q[2]; };

__device__ inline v16bf make_frag(const bf16_t* p0, const bf16_t* p1) {
  Frag16 f;
  f.q[0] = *reinterpret_cast<const uint4*>(p0);
  f.q[1] = *reinterpret_cast<const uint4*>(p1);
  return f.v;
}

// ---- Tensor Data Mover: DMA a [rows=16 x cols=64 bf16] 2D tile (row stride =
// strideElems) from global into LDS at lds_addr (row-major, contiguous 2KB).
// D# built per CDNA5 ISA 8.3/8.4; this toolchain exposes the 6-arg builtin:
// (u32x4 g0, i32x8 g1, i32x4 g2, i32x4 g3, i32x8 extra, i32 cpol).
__device__ inline void tdm_load_tile_16x64(unsigned lds_addr, const void* gptr,
                                           int strideElems) {
  unsigned long ga = (unsigned long)gptr;
  u32x4 g0;
  g0[0] = 1u;                                            // count=1, user mode
  g0[1] = lds_addr;                                      // lds_addr
  g0[2] = (unsigned)ga;                                  // global_addr[31:0]
  g0[3] = (unsigned)((ga >> 32) & 0x1FFFFFFu) | (2u << 30); // addr[56:32]|type=2
  i32x8 g1;
  g1[0] = (1 << 16);            // workgroup_mask=0, data_size=1 (2 bytes)
  g1[1] = 0;                    // tensor_dim0[15:0] in [31:16] -> dim0=0x40000000
  g1[2] = 0x4000;               // dim0[31:16]=0x4000 ; dim1[15:0]=0
  g1[3] = 0x4000 | (64 << 16);  // dim1[31:16]=0x4000 ; tile_dim0=64
  g1[4] = 16;                   // tile_dim1=16 ; tile_dim2=0
  g1[5] = strideElems;          // tensor_dim0_stride[31:0] (elements)
  g1[6] = 0;                    // stride hi / dim1_stride lo
  g1[7] = 0;
  i32x4 z4 = {0, 0, 0, 0};      // groups 2/3 unused (2D tensor)
  i32x8 z8 = {0, 0, 0, 0, 0, 0, 0, 0};
  __builtin_amdgcn_tensor_load_to_lds(g0, g1, z4, z4, z8, 0);
}

// ---- per-lane async gather of a [16 x 64 bf16] tile (row stride K) into LDS.
__device__ inline void async_load_tile_16x64(unsigned lds_addr, const bf16_t* src0,
                                             long K, int lane) {
  const int row = lane >> 1;
  const unsigned dbase = lds_addr + (unsigned)row * 128u;
  const bf16_t* srow = src0 + (long)row * K;
#pragma unroll
  for (int j = 0; j < 4; ++j) {
    const int seg = (lane & 1) + 2 * j;                  // 16B segment 0..7
    asm volatile("global_load_async_to_lds_b128 %0, %1, off"
                 :: "v"(dbase + (unsigned)seg * 16u), "v"(srow + seg * 8)
                 : "memory");
  }
}

// ---------------------------------------------------------------- cast f32 -> bf16
__global__ __launch_bounds__(256) void k_cast_bf16(const float* __restrict__ in,
                                                   bf16_t* __restrict__ out, long n) {
  long i = (long)blockIdx.x * blockDim.x + threadIdx.x;
  long stride = (long)gridDim.x * blockDim.x;
  for (; i < n; i += stride) out[i] = (bf16_t)in[i];
}

// ---------------------------------------------------------------- f[m][n] = sum_k A[m][k]*B[n][k]
// Both operands row-major [rows x K] bf16, contraction along the fast axis.
// Per-wave double-buffered staging: A tile via TDM, B tile via async loads,
// drained with the split s_wait_tensorcnt / s_wait_asynccnt counters; fragments
// then come from LDS as ds_load_b128 pairs (integer offsets into the __shared__
// array keep address-space inference intact). One wave per 16x16 output tile.
__global__ __launch_bounds__(256) void k_attn_f(
    const bf16_t* __restrict__ A, const bf16_t* __restrict__ B,
    float* __restrict__ F, int Msz, int K,
    long strideA, long strideB, long strideF) {
  __shared__ bf16_t smem[8 * 4096];  // per wave: A[2][1024] + B[2][1024] (8KB)
  const int wave = threadIdx.x >> 5, lane = threadIdx.x & 31;
  const int l15 = lane & 15;
  const int b = blockIdx.z;
  const bf16_t* Ab = A + (long)b * strideA;
  const bf16_t* Bb = B + (long)b * strideB;
  const int tiles_n = Msz >> 4;
  const int tile = blockIdx.x * 8 + wave;
  const int m0 = (tile / tiles_n) << 4;
  const int n0 = (tile % tiles_n) << 4;

  // element offsets into smem (keep all LDS accesses rooted at the shared array)
  const unsigned wbase = (unsigned)wave * 4096u;
  const unsigned aoff0 = wbase, aoff1 = wbase + 1024u;
  const unsigned boff0 = wbase + 2048u, boff1 = wbase + 3072u;

  v8f acc = {};
  int buf = 0;
  // prologue: stage chunk k0=0
  tdm_load_tile_16x64((unsigned)(unsigned long)&smem[aoff0],
                      Ab + (long)m0 * K, K);
  async_load_tile_16x64((unsigned)(unsigned long)&smem[boff0],
                        Bb + (long)n0 * K, K, lane);

  for (int k0 = 0; k0 < K; k0 += 64) {
    const bool more = (k0 + 64) < K;
    if (more) {  // stage next chunk into the other buffer
      tdm_load_tile_16x64((unsigned)(unsigned long)&smem[buf ? aoff0 : aoff1],
                          Ab + (long)m0 * K + (k0 + 64), K);
      async_load_tile_16x64((unsigned)(unsigned long)&smem[buf ? boff0 : boff1],
                            Bb + (long)n0 * K + (k0 + 64), K, lane);
      asm volatile("s_wait_tensorcnt 0x1" ::: "memory");
      asm volatile("s_wait_asynccnt 0x4" ::: "memory");
    } else {
      asm volatile("s_wait_tensorcnt 0x0" ::: "memory");
      asm volatile("s_wait_asynccnt 0x0" ::: "memory");
    }
    const unsigned ab = buf ? aoff1 : aoff0;
    const unsigned bb = buf ? boff1 : boff0;
    // A frag: lanes<16 k {0..7,16..23}; lanes>=16 k {8..15,24..31}
    const bf16_t* ap = &smem[ab + (unsigned)(l15 * 64) + ((lane < 16) ? 0u : 8u)];
    // B frag: lanes<16 k {0..7,16..23}; lanes>=16 k {8..15,24..31} (col = lane&15)
    const bf16_t* bp = &smem[bb + (unsigned)(l15 * 64) + ((lane < 16) ? 0u : 8u)];
#pragma unroll
    for (int ks = 0; ks < 64; ks += 32) {
      v16bf fa = make_frag(ap + ks, ap + ks + 16);
      v16bf fb = make_frag(bp + ks, bp + ks + 16);
      acc = __builtin_amdgcn_wmma_f32_16x16x32_bf16(false, fa, false, fb,
                                                    (short)0, acc, false, false);
    }
    buf ^= 1;
  }

  float* Fb = F + (long)b * strideF;
#pragma unroll
  for (int r = 0; r < 8; ++r)
    Fb[(long)(m0 + r + 8 * (lane >> 4)) * Msz + n0 + l15] = acc[r];
}

// ---------------------------------------------------------------- softmax over axis=1 (rows), per column
__global__ __launch_bounds__(128) void k_softmax_col(float* __restrict__ F, int Msz, long strideF) {
  const int col = threadIdx.x;
  if (col >= Msz) return;
  float* Fb = F + (long)blockIdx.z * strideF;
  float mx = -3.4e38f;
  for (int r = 0; r < Msz; ++r) mx = fmaxf(mx, Fb[(long)r * Msz + col]);
  float s = 0.f;
  for (int r = 0; r < Msz; ++r) s += __expf(Fb[(long)r * Msz + col] - mx);
  float inv = 1.0f / s;
  for (int r = 0; r < Msz; ++r) {
    long i = (long)r * Msz + col;
    Fb[i] = __expf(Fb[i] - mx) * inv;
  }
}

// ---------------------------------------------------------------- generic GEMM:
// Out[m][n] = sum_k A[m][k] * B[k][n]  (A f32 row-major MxK, B bf16 row-major KxN)
// M = MT*16.  8 waves; wave w owns 16-column subtile w of a 128-wide N tile.
// modes: 0 = flat bf16 (+bias,+residual)   1 = permuted bf16: out[n*M + m]
//        3 = f32 with bias + LeakyReLU(0.2)
template <int MT>
__global__ __launch_bounds__(256) void k_gemm_bf16(
    const float* __restrict__ A, const bf16_t* __restrict__ B,
    const float* __restrict__ bias, const float* __restrict__ res,
    void* __restrict__ Out, int K, int N,
    long strideA, long strideB, long strideR, long strideO, int mode) {
  constexpr int M = MT * 16;
  __shared__ bf16_t sA[M * 64];      // row-major [m][k]
  __shared__ bf16_t sB[128 * 64];    // transposed  [n][k]
  const int wave = threadIdx.x >> 5, lane = threadIdx.x & 31;
  const int l15 = lane & 15;
  const int b = blockIdx.z;
  const int n0 = blockIdx.x * 128;
  const float*  Ab = A + (long)b * strideA;
  const bf16_t* Bb = B + (long)b * strideB;

  v8f acc[MT];
#pragma unroll
  for (int mi = 0; mi < MT; ++mi) acc[mi] = {};

  for (int k0 = 0; k0 < K; k0 += 64) {
    for (int idx = threadIdx.x; idx < M * 64; idx += 256)
      sA[idx] = (bf16_t)Ab[(long)(idx >> 6) * K + k0 + (idx & 63)];
    for (int idx = threadIdx.x; idx < 64 * 16; idx += 256) {
      const int k = idx >> 4, nv = (idx & 15) << 3;
      uint4 d = *reinterpret_cast<const uint4*>(Bb + (long)(k0 + k) * N + n0 + nv);
      bf16_t tmp[8];
      __builtin_memcpy(tmp, &d, 16);
#pragma unroll
      for (int j = 0; j < 8; ++j) sB[(nv + j) * 64 + k] = tmp[j];
    }
    __syncthreads();
    const int ka = (lane < 16) ? 0 : 8;
    const int kb = (lane < 16) ? 0 : 8;
#pragma unroll
    for (int mi = 0; mi < MT; ++mi) {
      const bf16_t* ap = &sA[(mi * 16 + l15) * 64 + ka];
      const bf16_t* bp = &sB[(16 * wave + l15) * 64 + kb];
#pragma unroll
      for (int ks = 0; ks < 64; ks += 32) {
        v16bf fa = make_frag(ap + ks, ap + ks + 16);
        v16bf fb = make_frag(bp + ks, bp + ks + 16);
        acc[mi] = __builtin_amdgcn_wmma_f32_16x16x32_bf16(false, fa, false, fb,
                                                          (short)0, acc[mi], false, false);
      }
    }
    __syncthreads();
  }

  const int ncol = n0 + 16 * wave + l15;
  const float* resb = res ? res + (long)b * strideR : nullptr;
#pragma unroll
  for (int mi = 0; mi < MT; ++mi) {
#pragma unroll
    for (int r = 0; r < 8; ++r) {
      const int m = mi * 16 + r + 8 * (lane >> 4);
      float v = acc[mi][r];
      if (bias) v += bias[m];
      if (resb) v += resb[(long)m * N + ncol];
      if (mode == 3) {
        float* of = (float*)Out + (long)b * strideO;
        of[(long)m * N + ncol] = (v >= 0.f) ? v : 0.2f * v;
      } else if (mode == 1) {
        bf16_t* ob = (bf16_t*)Out + (long)b * strideO;
        ob[(long)ncol * M + m] = (bf16_t)v;
      } else {
        bf16_t* ob = (bf16_t*)Out + (long)b * strideO;
        ob[(long)m * N + ncol] = (bf16_t)v;
      }
    }
  }
}

// ---------------------------------------------------------------- host
extern "C" void kernel_launch(void* const* d_in, const int* in_sizes, int n_in,
                              void* d_out, int out_size, void* d_ws, size_t ws_size,
                              hipStream_t stream) {
  const float* x = (const float*)d_in[0];
  const float* gc_w  = (const float*)d_in[1];  const float* gc_b  = (const float*)d_in[2];
  const float* thc_w = (const float*)d_in[3];  const float* thc_b = (const float*)d_in[4];
  const float* phc_w = (const float*)d_in[5];  const float* phc_b = (const float*)d_in[6];
  const float* Wc_w  = (const float*)d_in[7];  const float* Wc_b  = (const float*)d_in[8];
  const float* gh_w  = (const float*)d_in[9];  const float* gh_b  = (const float*)d_in[10];
  const float* thh_w = (const float*)d_in[11]; const float* thh_b = (const float*)d_in[12];
  const float* phh_w = (const float*)d_in[13]; const float* phh_b = (const float*)d_in[14];
  const float* Wh_w  = (const float*)d_in[15]; const float* Wh_b  = (const float*)d_in[16];
  const float* gw_w  = (const float*)d_in[17]; const float* gw_b  = (const float*)d_in[18];
  const float* thw_w = (const float*)d_in[19]; const float* thw_b = (const float*)d_in[20];
  const float* phw_w = (const float*)d_in[21]; const float* phw_b = (const float*)d_in[22];
  const float* Ww_w  = (const float*)d_in[23]; const float* Ww_b  = (const float*)d_in[24];
  const float* fus_w = (const float*)d_in[25]; const float* fus_b = (const float*)d_in[26];

  const long CH = 64L * 16384;   // per-batch tensor elements
  const long PB = 14 * CH;       // full tensor elements

  bf16_t* xb = (bf16_t*)d_ws;
  bf16_t* gS = xb + PB;
  bf16_t* tS = gS + PB;
  bf16_t* pS = tS + PB;
  bf16_t* yS = pS + PB;
  bf16_t* zS = yS + PB;                 // 3*PB (zc | zh | zw)
  float*  fS = (float*)(zS + 3 * PB);   // attention matrices (<= 14*128*128 f32)

  k_cast_bf16<<<14336, 256, 0, stream>>>(x, xb, PB);

  struct Br { const float *gw,*gb,*tw,*tb,*pw,*pb,*Ww,*Wb; int hw; };
  Br brs[3] = {
    { gc_w, gc_b, thc_w, thc_b, phc_w, phc_b, Wc_w, Wc_b, 0 },
    { gh_w, gh_b, thh_w, thh_b, phh_w, phh_b, Wh_w, Wh_b, 1 },
    { gw_w, gw_b, thw_w, thw_b, phw_w, phw_b, Ww_w, Ww_b, 1 },
  };

  for (int br = 0; br < 3; ++br) {
    const Br& s = brs[br];
    k_gemm_bf16<4><<<dim3(128, 1, 14), 256, 0, stream>>>(
        s.gw, xb, s.gb, nullptr, gS, 64, 16384, 0, CH, 0, CH, 0);
    k_gemm_bf16<4><<<dim3(128, 1, s.hw ? 1 : 14), 256, 0, stream>>>(
        s.tw, xb, s.tb, nullptr, tS, 64, 16384, 0, CH, 0, CH, 0);
    k_gemm_bf16<4><<<dim3(128, 1, 14), 256, 0, stream>>>(
        s.pw, xb, s.pb, nullptr, pS, 64, 16384, 0, CH, 0, CH, 0);

    if (!s.hw) {
      k_attn_f<<<dim3(2, 1, 14), 256, 0, stream>>>(tS, pS, fS, 64, 16384, CH, CH, 64 * 64);
      k_softmax_col<<<dim3(1, 1, 14), 128, 0, stream>>>(fS, 64, 64 * 64);
      k_gemm_bf16<4><<<dim3(128, 1, 14), 256, 0, stream>>>(
          fS, gS, nullptr, nullptr, yS, 64, 16384, 64 * 64, CH, 0, CH, 1);
    } else {
      k_attn_f<<<dim3(8, 1, 14), 256, 0, stream>>>(tS, pS, fS, 128, 8192, 0, CH, 128 * 128);
      k_softmax_col<<<dim3(1, 1, 14), 128, 0, stream>>>(fS, 128, 128 * 128);
      k_gemm_bf16<8><<<dim3(64, 1, 14), 256, 0, stream>>>(
          fS, gS, nullptr, nullptr, yS, 128, 8192, 128 * 128, CH, 0, CH, 1);
    }
    k_gemm_bf16<4><<<dim3(128, 1, 14), 256, 0, stream>>>(
        s.Ww, yS, s.Wb, x, zS + (long)br * PB, 64, 16384, 0, CH, CH, CH, 0);
  }

  k_gemm_bf16<4><<<dim3(128, 1, 1), 256, 0, stream>>>(
      fus_w, zS, fus_b, nullptr, d_out, 2688, 16384, 0, 0, 0, 0, 3);

  (void)in_sizes; (void)n_in; (void)out_size; (void)ws_size;
}
